// BigramModel_50594714747029
// MI455X (gfx1250) — matrix operand order repaired and checked by reading, not compiled, not run on hardware
//
#include <hip/hip_runtime.h>

typedef __bf16 bf16;
typedef __attribute__((ext_vector_type(16))) __bf16 v16bf;
typedef __attribute__((ext_vector_type(8)))  __bf16 v8bf;
typedef __attribute__((ext_vector_type(8)))  float  v8f;
typedef unsigned int v4u __attribute__((ext_vector_type(4)));
typedef int          v8i __attribute__((ext_vector_type(8)));
typedef int          v4i __attribute__((ext_vector_type(4)));

// Problem dims are fixed by the reference — use shifts, not div/mod.
#define HD    64     // head size
#define TDIM  1024   // sequence length (shift 10)
#define HDIM  16     // heads           (shift 4)
#define EDIM  1024   // embed           (shift 10)

// ---------------------------------------------------------------------------
// WMMA helpers (CDNA5 gfx1250, wave32)
// ---------------------------------------------------------------------------
static __device__ __forceinline__ v8f wmma_bf16(v16bf a, v16bf b, v8f c) {
  return __builtin_amdgcn_wmma_f32_16x16x32_bf16(
      false, a, false, b, (short)0, c, false, false);
}

// A-matrix 16x32 bf16 fragment from row-major storage (ld in halfs).
static __device__ __forceinline__ v16bf load_a_frag(const bf16* base, int ld) {
  int l  = threadIdx.x & 31;
  int r  = l & 15;
  int kb = (l >> 4) << 3;
  const bf16* p = base + r * ld;
  union { v16bf v; v8bf h[2]; } u;
  u.h[0] = *(const v8bf*)(p + kb);
  u.h[1] = *(const v8bf*)(p + 16 + kb);
  return u.v;
}

// B-matrix 32x16 bf16 fragment from TRANSPOSED storage base[n*ld + k].
static __device__ __forceinline__ v16bf load_b_frag(const bf16* base, int ld) {
  int l  = threadIdx.x & 31;
  int n  = l & 15;
  int kb = (l >> 4) << 3;
  const bf16* p = base + n * ld + kb;
  union { v16bf v; v8bf h[2]; } u;
  u.h[0] = *(const v8bf*)p;
  u.h[1] = *(const v8bf*)(p + 16);
  return u.v;
}

// ---------------------------------------------------------------------------
// Tensor Data Mover: 2-D tile load (global -> LDS) with LDS row padding.
// D# built per CDNA5 ISA 08_async_tensor §8.3/§8.4.
// LDS rows padded: pad_interval code 3 (16 DWORDs) + pad_amount code 3 (4 DW),
// matching a [rows][40] bf16 LDS array when tileK == 32.
// ---------------------------------------------------------------------------
#if __clang_major__ >= 23
#define TDM_LOAD(g0, g1, gz) \
  __builtin_amdgcn_tensor_load_to_lds((g0), (g1), (gz), (gz), \
                                      (v8i){0,0,0,0,0,0,0,0}, 0)
#else
#define TDM_LOAD(g0, g1, gz) \
  __builtin_amdgcn_tensor_load_to_lds((g0), (g1), (gz), (gz), 0)
#endif

static __device__ __forceinline__ v8i tdm_make_g1(int K, int tileK, int tileRows,
                                                  int rowsAvail) {
  v8i g1;
  // w0: wg_mask=0 | data_size=1(2B)<<16 | pad_enable<<20 | pad_interval=3<<22
  //     | pad_amount=3<<25
  g1[0] = (1 << 16) | (1 << 20) | (3 << 22) | (3 << 25);
  g1[1] = (int)((unsigned)K << 16);            // tensor_dim0[15:0] @ bits 63:48
  g1[2] = (int)(((unsigned)K >> 16) | ((unsigned)rowsAvail << 16)); // td0 hi | td1 lo
  g1[3] = (int)(((unsigned)rowsAvail >> 16) | ((unsigned)tileK << 16)); // td1 hi | tile_dim0
  g1[4] = tileRows;                            // tile_dim1 (tile_dim2 = 0)
  g1[5] = K;                                   // tensor_dim0_stride[31:0]
  g1[6] = 0;
  g1[7] = 0;
  return g1;
}

static __device__ __forceinline__ v4u tdm_make_g0(unsigned lds_off, unsigned long long ga) {
  v4u g0;
  g0[0] = 1u;                                  // count=1, user mode
  g0[1] = lds_off;                             // lds_addr
  g0[2] = (unsigned)ga;                        // global_addr[31:0]
  g0[3] = (unsigned)((ga >> 32) & 0x1FFFFFFu) | 0x80000000u; // ga[56:32] | type=2
  return g0;
}

// ---------------------------------------------------------------------------
// 1) f32 -> bf16 conversion (weights)
// ---------------------------------------------------------------------------
__global__ void cvt_bf16_kernel(const float* __restrict__ s, bf16* __restrict__ d,
                                size_t n) {
  size_t i  = (size_t)blockIdx.x * blockDim.x + threadIdx.x;
  size_t st = (size_t)gridDim.x * blockDim.x;
  for (; i < n; i += st) d[i] = (bf16)s[i];
}

// ---------------------------------------------------------------------------
// 2) embedding: x[b,t,:] = tok_emb[ids[b,t],:] + pos_emb[t,:]  (bf16 out)
// ---------------------------------------------------------------------------
__global__ void embed_kernel(const int* __restrict__ ids,
                             const float* __restrict__ tok,
                             const float* __restrict__ pos,
                             bf16* __restrict__ xo, size_t total) {
  size_t i  = (size_t)blockIdx.x * blockDim.x + threadIdx.x;
  size_t st = (size_t)gridDim.x * blockDim.x;
  for (; i < total; i += st) {
    size_t e  = i & (EDIM - 1);
    size_t bt = i >> 10;                 // / EDIM
    size_t t  = bt & (TDIM - 1);
    int id    = ids[bt];
    xo[i] = (bf16)(tok[(size_t)id * EDIM + e] + pos[t * EDIM + e]);
  }
}

// ---------------------------------------------------------------------------
// 3) QKV projection: out[b,h,t,:] = x[b,t,:] @ W[h]   (K=E, N=HD=64)
//    grid = (M/64, H), 128 threads (4 waves)
// ---------------------------------------------------------------------------
__global__ __launch_bounds__(128) void qkv_kernel(const bf16* __restrict__ X,
                                                  const bf16* __restrict__ W,
                                                  bf16* __restrict__ out, int K) {
  __shared__ bf16 lds_B[HD][40];     // [n][k], +8 halfs pad
  int h   = blockIdx.y;
  int m0  = blockIdx.x * 64;
  int tid = threadIdx.x, wid = tid >> 5, lane = tid & 31;
  const bf16* Wh = W + (size_t)h * K * HD;

  v8f acc[4] = {};
  int r   = lane & 15;
  int kb8 = (lane >> 4) << 3;
  const bf16* arow = X + (size_t)(m0 + wid * 16 + r) * K;

  for (int k0 = 0; k0 < K; k0 += 32) {
#pragma unroll
    for (int i = 0; i < 2; ++i) {
      int c  = tid + i * 128;
      int kr = c >> 3, cc = (c & 7) << 3;
      v8bf wv = *(const v8bf*)(Wh + (size_t)(k0 + kr) * HD + cc);
#pragma unroll
      for (int j = 0; j < 8; ++j) lds_B[cc + j][kr] = wv[j];
    }
    __syncthreads();

    union { v16bf v; v8bf h2[2]; } ua;
    ua.h2[0] = *(const v8bf*)(arow + k0 + kb8);
    ua.h2[1] = *(const v8bf*)(arow + k0 + 16 + kb8);
#pragma unroll
    for (int nt = 0; nt < 4; ++nt) {
      v16bf bfv = load_b_frag(&lds_B[nt * 16][0], 40);
      acc[nt] = wmma_bf16(ua.v, bfv, acc[nt]);
    }
    __syncthreads();
  }

  int half = lane >> 4, nlo = lane & 15;
#pragma unroll
  for (int nt = 0; nt < 4; ++nt)
#pragma unroll
    for (int j = 0; j < 8; ++j) {
      int m = m0 + wid * 16 + j + 8 * half;
      int b = m >> 10;                   // / TDIM
      int t = m & (TDIM - 1);
      out[((size_t)((b << 4) + h) * TDIM + t) * HD + nt * 16 + nlo] = (bf16)acc[nt][j];
    }
}

// ---------------------------------------------------------------------------
// 4) Flash attention (wei = k @ q^T, causal s<=t, softmax, @v)
//    grid = (T/128, B*H), 256 threads; wave w owns rows r0 = m0+16w.
// ---------------------------------------------------------------------------
__global__ __launch_bounds__(256) void attn_kernel(const bf16* __restrict__ kmat,
                                                   const bf16* __restrict__ qmat,
                                                   const bf16* __restrict__ vmat,
                                                   bf16* __restrict__ z) {
  __shared__ bf16 lds_vT[HD][40];      // [d][kk]
  __shared__ bf16 lds_p[8][16][40];    // per-wave P scratch [row][kk]

  int bh  = blockIdx.y;
  int b   = bh >> 4;                   // / HDIM
  int h   = bh & (HDIM - 1);
  int m0  = blockIdx.x * 128;
  int tid = threadIdx.x, wid = tid >> 5, lane = tid & 31;
  int r0  = m0 + wid * 16;
  int half = lane >> 4, nlo = lane & 15;

  const bf16* kbase = kmat + (size_t)bh * TDIM * HD;
  const bf16* qbase = qmat + (size_t)bh * TDIM * HD;
  const bf16* vbase = vmat + (size_t)bh * TDIM * HD;

  v16bf aK0, aK1;
  {
    int r  = lane & 15;
    int kb = (lane >> 4) << 3;
    const bf16* p = kbase + (size_t)(r0 + r) * HD;
    union { v16bf v; v8bf h2[2]; } u0, u1;
    u0.h2[0] = *(const v8bf*)(p + kb);
    u0.h2[1] = *(const v8bf*)(p + 16 + kb);
    u1.h2[0] = *(const v8bf*)(p + 32 + kb);
    u1.h2[1] = *(const v8bf*)(p + 48 + kb);
    aK0 = u0.v; aK1 = u1.v;
  }

  v8f o[4] = {};
  float mst[8], lst[8];
#pragma unroll
  for (int j = 0; j < 8; ++j) { mst[j] = -1e30f; lst[j] = 0.f; }

  int nblk = (m0 >> 5) + 4;
  for (int blk = 0; blk < nblk; ++blk) {
    int s0 = blk * 32;

    {   // cooperative stage: vT[d][kk] = v[s0+kk][d]
      int kk = tid >> 3;
      int c0 = (tid & 7) * 8;
      v8bf vv = *(const v8bf*)(vbase + (size_t)(s0 + kk) * HD + c0);
#pragma unroll
      for (int j = 0; j < 8; ++j) lds_vT[c0 + j][kk] = vv[j];
    }
    __syncthreads();

    v8f sv[2];
#pragma unroll
    for (int nt = 0; nt < 2; ++nt) {
      v8f acc = {};
#pragma unroll
      for (int kh = 0; kh < 2; ++kh) {
        const bf16* p = qbase + (size_t)(s0 + nt * 16 + nlo) * HD + kh * 32 + (half << 3);
        union { v16bf v; v8bf h2[2]; } u;
        u.h2[0] = *(const v8bf*)p;
        u.h2[1] = *(const v8bf*)(p + 16);
        acc = wmma_bf16(kh ? aK1 : aK0, u.v, acc);
      }
      sv[nt] = acc;
    }

#pragma unroll
    for (int j = 0; j < 8; ++j) {
      int t = r0 + j + 8 * half;
#pragma unroll
      for (int nt = 0; nt < 2; ++nt) {
        int sc = s0 + nt * 16 + nlo;
        float x = sv[nt][j] * 0.125f;
        sv[nt][j] = (sc <= t) ? x : -1e30f;
      }
    }

#pragma unroll
    for (int j = 0; j < 8; ++j) {
      float rm = fmaxf(sv[0][j], sv[1][j]);
      rm = fmaxf(rm, __shfl_xor(rm, 1, 32));
      rm = fmaxf(rm, __shfl_xor(rm, 2, 32));
      rm = fmaxf(rm, __shfl_xor(rm, 4, 32));
      rm = fmaxf(rm, __shfl_xor(rm, 8, 32));
      float mnew = fmaxf(mst[j], rm);
      float corr = __expf(mst[j] - mnew);
      mst[j] = mnew;

      float p0 = __expf(sv[0][j] - mnew);
      float p1 = __expf(sv[1][j] - mnew);
      int row = j + 8 * half;
      lds_p[wid][row][nlo]      = (bf16)p0;
      lds_p[wid][row][16 + nlo] = (bf16)p1;

      float rs = p0 + p1;
      rs += __shfl_xor(rs, 1, 32);
      rs += __shfl_xor(rs, 2, 32);
      rs += __shfl_xor(rs, 4, 32);
      rs += __shfl_xor(rs, 8, 32);
      lst[j] = lst[j] * corr + rs;
#pragma unroll
      for (int nt = 0; nt < 4; ++nt) o[nt][j] *= corr;
    }

    v16bf aP = load_a_frag(&lds_p[wid][0][0], 40);
#pragma unroll
    for (int nt = 0; nt < 4; ++nt) {
      v16bf bv = load_b_frag(&lds_vT[nt * 16][0], 40);
      o[nt] = wmma_bf16(aP, bv, o[nt]);
    }
    __syncthreads();
  }

#pragma unroll
  for (int nt = 0; nt < 4; ++nt)
#pragma unroll
    for (int j = 0; j < 8; ++j) {
      int row = j + 8 * half;
      float val = o[nt][j] / lst[j];
      z[(size_t)(b * TDIM + r0 + row) * EDIM + h * HD + nt * 16 + nlo] = (bf16)val;
    }
}

// ---------------------------------------------------------------------------
// 5) 128x128 GEMM, software-pipelined with the Tensor Data Mover:
//    double-buffered LDS tiles; wave 0 issues the TDM for tile i+1 and all
//    threads issue global B loads for tile i+1 BEFORE the 8 WMMAs of tile i,
//    so the DMA and global loads overlap matrix math. One barrier / iter.
// ---------------------------------------------------------------------------
__global__ __launch_bounds__(256) void gemm_bf16_128x128(const bf16* __restrict__ A,
                                                         const bf16* __restrict__ Bm,
                                                         const float* __restrict__ bias,
                                                         float* __restrict__ Cf,
                                                         bf16* __restrict__ Cb,
                                                         int M, int N, int K, int relu) {
  __shared__ bf16 lds_A[2][128][40];   // row-major, +8 pad (TDM pads 4DW/16DW)
  __shared__ bf16 lds_B[2][128][40];   // transposed [n][k], +8 pad
  int n0  = blockIdx.x * 128;
  int m0  = blockIdx.y * 128;
  int tid = threadIdx.x, wid = tid >> 5, lane = tid & 31;
  int wm  = (wid >> 2) * 64;
  int wn  = (wid & 3) * 32;

  v8f acc[4][2] = {};

  unsigned ldsA_off[2] = { (unsigned)(uintptr_t)(&lds_A[0][0][0]),
                           (unsigned)(uintptr_t)(&lds_A[1][0][0]) };
  unsigned long long gaBase = (unsigned long long)(uintptr_t)(A + (size_t)m0 * K);
  v8i g1 = tdm_make_g1(K, 32, 128, 128);
  v4i gz = {0, 0, 0, 0};

  // per-thread B-staging coordinates (two 8-half chunks of the 32x128 tile)
  int kr0 = tid >> 4,        cc0 = (tid & 15) << 3;
  int kr1 = (tid + 256) >> 4, cc1 = ((tid + 256) & 15) << 3;

  // ---- prologue: tile 0 into buffer 0
  if (wid == 0) {
    v4u g0 = tdm_make_g0(ldsA_off[0], gaBase);
    TDM_LOAD(g0, g1, gz);
  }
  v8bf b0 = *(const v8bf*)(Bm + (size_t)kr0 * N + n0 + cc0);
  v8bf b1 = *(const v8bf*)(Bm + (size_t)kr1 * N + n0 + cc1);
#pragma unroll
  for (int j = 0; j < 8; ++j) lds_B[0][cc0 + j][kr0] = b0[j];
#pragma unroll
  for (int j = 0; j < 8; ++j) lds_B[0][cc1 + j][kr1] = b1[j];
  if (wid == 0) __builtin_amdgcn_s_wait_tensorcnt(0);
  __syncthreads();

  int nk = K >> 5;
  for (int i = 0; i < nk; ++i) {
    int cur = i & 1, nxt = cur ^ 1;
    int kn  = (i + 1) << 5;
    bool more = kn < K;

    if (more) {   // issue next tile's DMA + global loads before computing
      if (wid == 0) {
        v4u g0 = tdm_make_g0(ldsA_off[nxt], gaBase + (unsigned long long)kn * 2u);
        TDM_LOAD(g0, g1, gz);
      }
      b0 = *(const v8bf*)(Bm + (size_t)(kn + kr0) * N + n0 + cc0);
      b1 = *(const v8bf*)(Bm + (size_t)(kn + kr1) * N + n0 + cc1);
    }

    // compute tile i from the current buffers
    v16bf af[4];
#pragma unroll
    for (int mt = 0; mt < 4; ++mt)
      af[mt] = load_a_frag(&lds_A[cur][wm + mt * 16][0], 40);
#pragma unroll
    for (int nt = 0; nt < 2; ++nt) {
      v16bf bfv = load_b_frag(&lds_B[cur][wn + nt * 16][0], 40);
#pragma unroll
      for (int mt = 0; mt < 4; ++mt)
        acc[mt][nt] = wmma_bf16(af[mt], bfv, acc[mt][nt]);
    }

    if (more) {   // publish next B tile, close next A DMA, single barrier
#pragma unroll
      for (int j = 0; j < 8; ++j) lds_B[nxt][cc0 + j][kr0] = b0[j];
#pragma unroll
      for (int j = 0; j < 8; ++j) lds_B[nxt][cc1 + j][kr1] = b1[j];
      if (wid == 0) __builtin_amdgcn_s_wait_tensorcnt(0);
      __syncthreads();
    }
  }

  int half = lane >> 4, nlo = lane & 15;
#pragma unroll
  for (int nt = 0; nt < 2; ++nt) {
    int col = n0 + wn + nt * 16 + nlo;
    float bv = bias ? bias[col] : 0.f;
#pragma unroll
    for (int mt = 0; mt < 4; ++mt)
#pragma unroll
      for (int j = 0; j < 8; ++j) {
        int row = m0 + wm + mt * 16 + j + 8 * half;
        float v = acc[mt][nt][j] + bv;
        if (relu) v = fmaxf(v, 0.f);
        if (Cf) Cf[(size_t)row * N + col] = v;
        if (Cb) Cb[(size_t)row * N + col] = (bf16)v;
      }
  }
}

// ---------------------------------------------------------------------------
// host launcher
// ---------------------------------------------------------------------------
extern "C" void kernel_launch(void* const* d_in, const int* in_sizes, int n_in,
                              void* d_out, int out_size, void* d_ws, size_t ws_size,
                              hipStream_t stream) {
  (void)in_sizes; (void)n_in; (void)out_size; (void)ws_size;
  const int Bc = 4, Tc = 1024, Vc = 32000, Ec = 1024, Hc = 16;
  const int Mr = Bc * Tc;  // 4096

  const int*   ids  = (const int*)d_in[0];
  const float* tok  = (const float*)d_in[1];
  const float* pos  = (const float*)d_in[2];
  const float* Wk   = (const float*)d_in[3];
  const float* Wq   = (const float*)d_in[4];
  const float* Wv   = (const float*)d_in[5];
  const float* ffW  = (const float*)d_in[6];
  const float* ffb  = (const float*)d_in[7];
  const float* outW = (const float*)d_in[8];
  const float* outb = (const float*)d_in[9];
  float* logits = (float*)d_out;

  char* ws = (char*)d_ws;
  size_t off = 0;
  auto take = [&](size_t elems) -> bf16* {
    bf16* p = (bf16*)(ws + off);
    off += ((elems * sizeof(bf16)) + 255) & ~(size_t)255;
    return p;
  };
  bf16* x_bf   = take((size_t)Mr * Ec);
  bf16* wk_bf  = take((size_t)Hc * Ec * HD);
  bf16* wq_bf  = take((size_t)Hc * Ec * HD);
  bf16* wv_bf  = take((size_t)Hc * Ec * HD);
  bf16* ffw_bf = take((size_t)Ec * Ec);
  bf16* ow_bf  = take((size_t)Ec * Vc);
  bf16* k_bf   = take((size_t)Bc * Hc * Tc * HD);
  bf16* q_bf   = take((size_t)Bc * Hc * Tc * HD);
  bf16* v_bf   = take((size_t)Bc * Hc * Tc * HD);
  bf16* z_bf   = take((size_t)Mr * Ec);
  bf16* z2_bf  = take((size_t)Mr * Ec);

  cvt_bf16_kernel<<<1024, 256, 0, stream>>>(Wk,   wk_bf,  (size_t)Hc * Ec * HD);
  cvt_bf16_kernel<<<1024, 256, 0, stream>>>(Wq,   wq_bf,  (size_t)Hc * Ec * HD);
  cvt_bf16_kernel<<<1024, 256, 0, stream>>>(Wv,   wv_bf,  (size_t)Hc * Ec * HD);
  cvt_bf16_kernel<<<1024, 256, 0, stream>>>(ffW,  ffw_bf, (size_t)Ec * Ec);
  cvt_bf16_kernel<<<2048, 256, 0, stream>>>(outW, ow_bf,  (size_t)Ec * Vc);

  embed_kernel<<<2048, 256, 0, stream>>>(ids, tok, pos, x_bf, (size_t)Mr * Ec);

  {
    dim3 g(Mr / 64, Hc);
    qkv_kernel<<<g, 128, 0, stream>>>(x_bf, wk_bf, k_bf, Ec);
    qkv_kernel<<<g, 128, 0, stream>>>(x_bf, wq_bf, q_bf, Ec);
    qkv_kernel<<<g, 128, 0, stream>>>(x_bf, wv_bf, v_bf, Ec);
  }

  {
    dim3 g(Tc / 128, Bc * Hc);
    attn_kernel<<<g, 256, 0, stream>>>(k_bf, q_bf, v_bf, z_bf);
  }

  {
    dim3 g(Ec / 128, Mr / 128);
    gemm_bf16_128x128<<<g, 256, 0, stream>>>(z_bf, ffw_bf, ffb,
                                             nullptr, z2_bf, Mr, Ec, Ec, 1);
  }

  {
    dim3 g(Vc / 128, Mr / 128);
    gemm_bf16_128x128<<<g, 256, 0, stream>>>(z2_bf, ow_bf, outb,
                                             logits, nullptr, Mr, Vc, Ec, 0);
  }
}
